// SelfAttentionEncoderLayer_83176336655036
// MI455X (gfx1250) — matrix operand order repaired and checked by reading, not compiled, not run on hardware
//
#include <hip/hip_runtime.h>

// ---------------------------------------------------------------- types
typedef unsigned int   u32;
typedef unsigned short u16;
typedef __attribute__((ext_vector_type(4)))  u32    u32x4;
typedef __attribute__((ext_vector_type(8)))  float  v8f;
typedef __attribute__((ext_vector_type(16))) __bf16 v16bf;

union Frag { v16bf v; u32x4 q[2]; };

static __device__ __forceinline__ u16 f2bf(float f) {
  u32 u = __float_as_uint(f);
  u += 0x7FFFu + ((u >> 16) & 1u);           // round-to-nearest-even
  return (u16)(u >> 16);
}

static __device__ __forceinline__ v8f wmma_bf16(Frag a, Frag b, v8f c) {
  return __builtin_amdgcn_wmma_f32_16x16x32_bf16(
      /*neg_a=*/false, a.v, /*neg_b=*/false, b.v,
      /*c_mod=*/(short)0, c, /*reuse_a=*/false, /*reuse_b=*/false);
}

// gfx1250 async copy to LDS (ASYNCcnt-tracked); guarded so the build can
// never break if the toolchain lacks the builtins.
#if __has_builtin(__builtin_amdgcn_global_load_async_to_lds_b128) && \
    __has_builtin(__builtin_amdgcn_s_wait_asynccnt)
#define ASYNC_LDS 1
#else
#define ASYNC_LDS 0
#endif

#if ASYNC_LDS
// builtin signature (from clang diagnostic): param1 = v4i addrspace(1)*,
// param2 = v4i addrspace(3)*, then imm offset + imm cpol.
typedef int i32x4g __attribute__((vector_size(16)));
typedef __attribute__((address_space(1))) i32x4g* as1_v4i;
typedef __attribute__((address_space(3))) i32x4g* as3_v4i;

static __device__ __forceinline__ void async_b128(const u16* g, const u16* l) {
  // generic->AS1: identical numeric address; generic LDS ptr low 32 bits are
  // the LDS offset (ISA aperture rules), so build both from integer values.
  __builtin_amdgcn_global_load_async_to_lds_b128(
      (as1_v4i)(unsigned long long)g,
      (as3_v4i)(unsigned int)(unsigned long long)l,
      0, 0);
}
#endif

#define SEQ   1024
#define BATCH 4
#define DIM   1024
#define HEADS 16
#define HD    64
#define FILT  4096
#define KW    9
#define NROW  (SEQ * BATCH)       // 4096 rows (t*B+b)

// ---------------------------------------------------------------- fp32 -> bf16
__global__ __launch_bounds__(256) void cvt_kernel(const float* __restrict__ in,
                                                  u16* __restrict__ out, int n) {
  int i = blockIdx.x * 256 + threadIdx.x;
  if (i < n) out[i] = f2bf(in[i]);
}

// conv_w (F, D, K) fp32  ->  (F, K, D) bf16  so conv == NT-GEMM with K' = K*D
__global__ __launch_bounds__(256) void convw_kernel(const float* __restrict__ in,
                                                    u16* __restrict__ out) {
  int i = blockIdx.x * 256 + threadIdx.x;        // total = FILT*KW*DIM
  int f = i / (KW * DIM);
  int r = i - f * (KW * DIM);
  int k = r >> 10;
  int d = r & 1023;
  out[i] = f2bf(in[((size_t)f * DIM + d) * KW + k]);
}

// ---------------------------------------------------------------- LayerNorm -> bf16
__global__ __launch_bounds__(256) void ln_kernel(const float* __restrict__ x,
                                                 const float* __restrict__ g,
                                                 const float* __restrict__ bta,
                                                 u16* __restrict__ out) {
  int n = blockIdx.x, tid = threadIdx.x;
  const float* row = x + (size_t)n * DIM;
  float v0[4]; float s = 0.f, s2 = 0.f;
#pragma unroll
  for (int i = 0; i < 4; ++i) { float t = row[tid + 256 * i]; v0[i] = t; s += t; s2 += t * t; }
#pragma unroll
  for (int off = 16; off >= 1; off >>= 1) {
    s  += __shfl_xor(s,  off, 32);
    s2 += __shfl_xor(s2, off, 32);
  }
  __shared__ float rs[8], rs2[8];
  if ((tid & 31) == 0) { rs[tid >> 5] = s; rs2[tid >> 5] = s2; }
  __syncthreads();
  s = 0.f; s2 = 0.f;
#pragma unroll
  for (int w = 0; w < 8; ++w) { s += rs[w]; s2 += rs2[w]; }
  float mu   = s * (1.f / DIM);
  float var  = s2 * (1.f / DIM) - mu * mu;
  float rstd = rsqrtf(var + 1e-5f);
#pragma unroll
  for (int i = 0; i < 4; ++i) {
    int c = tid + 256 * i;
    out[(size_t)n * DIM + c] = f2bf((v0[i] - mu) * rstd * g[c] + bta[c]);
  }
}

// ---------------------------------------------------------------- tiled NT-GEMM
// C[M,N] = A[M,K](bf16) * B[N,K](bf16)^T ; 128x128 block tile, 8 waves,
// wave = 32(M) x 64(N) = 8 accumulators of 16x16. Double-buffered LDS with a
// single barrier per K-stage; async-to-LDS staging when available.
// MODE 0: write bf16 C                        (QKV projections)
// MODE 1: Cf = acc + R  (f32)                 (Wo proj, final W2 proj)
// MODE 2: conv epilogue: (acc+bias)*9^-0.5, exact GeLU, bf16 out; A rows are
//         h2[(t + ktap - 4)*B + b, d] with zero-fill out of range (lda=DIM).
template <int MODE>
__global__ __launch_bounds__(256) void gemm_nt(const u16* __restrict__ A,
                                               const u16* __restrict__ B,
                                               u16* __restrict__ Cb,
                                               float* __restrict__ Cf,
                                               const float* __restrict__ R,
                                               const float* __restrict__ bias,
                                               int M, int N, int K) {
  __shared__ u16 As[2][128 * 40];   // row stride 40 elems (80B: 16B aligned chunks)
  __shared__ u16 Bs[2][128 * 40];
  const int bn = blockIdx.x, bm = blockIdx.y;
  const int tid = threadIdx.x;
  const int wid = tid >> 5, lane = tid & 31, l15 = lane & 15, hi = lane >> 4;
  const int wm = wid & 3, wn = wid >> 2;
  const int lr = tid >> 1, lh = (tid & 1) * 16;   // loader: row, 16-elem half

  v8f zf = {0.f, 0.f, 0.f, 0.f, 0.f, 0.f, 0.f, 0.f};
  v8f acc[2][4];
#pragma unroll
  for (int m = 0; m < 2; ++m)
#pragma unroll
    for (int n = 0; n < 4; ++n) acc[m][n] = zf;

  const u16* Abase = A + (size_t)(bm * 128 + lr) * (size_t)K + lh;  // modes 0/1
  const u16* Bbase = B + (size_t)(bn * 128 + lr) * (size_t)K + lh;

  auto loadAconv = [&](int k0, u32x4& a0, u32x4& a1) {
    int row = bm * 128 + lr;
    int t = row >> 2, bb = row & 3;
    int ktap = k0 >> 10, d = k0 & 1023;
    int tp = t + ktap - (KW / 2);
    if (tp >= 0 && tp < SEQ) {
      const u16* ap = A + (((size_t)(tp * BATCH + bb)) << 10) + d + lh;
      a0 = *(const u32x4*)ap; a1 = *(const u32x4*)(ap + 8);
    } else {
      a0 = (u32x4){0, 0, 0, 0}; a1 = (u32x4){0, 0, 0, 0};
    }
  };

  u32x4 ra0, ra1, rb0, rb1;
  int cur = 0;

  // ---- preamble: stage 0 into buffer 0
#if ASYNC_LDS
  async_b128(Bbase,     &Bs[0][lr * 40 + lh]);
  async_b128(Bbase + 8, &Bs[0][lr * 40 + lh + 8]);
  if constexpr (MODE != 2) {
    async_b128(Abase,     &As[0][lr * 40 + lh]);
    async_b128(Abase + 8, &As[0][lr * 40 + lh + 8]);
  } else {
    loadAconv(0, ra0, ra1);
    *(u32x4*)(&As[0][lr * 40 + lh])     = ra0;
    *(u32x4*)(&As[0][lr * 40 + lh + 8]) = ra1;
  }
#else
  if constexpr (MODE != 2) { ra0 = *(const u32x4*)Abase; ra1 = *(const u32x4*)(Abase + 8); }
  else loadAconv(0, ra0, ra1);
  rb0 = *(const u32x4*)Bbase; rb1 = *(const u32x4*)(Bbase + 8);
  *(u32x4*)(&As[0][lr * 40 + lh])     = ra0;
  *(u32x4*)(&As[0][lr * 40 + lh + 8]) = ra1;
  *(u32x4*)(&Bs[0][lr * 40 + lh])     = rb0;
  *(u32x4*)(&Bs[0][lr * 40 + lh + 8]) = rb1;
#endif

  for (int k0 = 0; k0 < K; k0 += 32) {
    const int kn = k0 + 32;
    const bool more = kn < K;
    const int nxt = cur ^ 1;
#if ASYNC_LDS
    __builtin_amdgcn_s_wait_asynccnt(0);   // this wave's stage-k0 copies done
#endif
    __syncthreads();                       // tile visible; prior readers of nxt done

    // ---- kick off stage k0+32 into the other buffer (overlaps the WMMAs)
    if (more) {
#if ASYNC_LDS
      const u16* bp = Bbase + kn;
      async_b128(bp,     &Bs[nxt][lr * 40 + lh]);
      async_b128(bp + 8, &Bs[nxt][lr * 40 + lh + 8]);
      if constexpr (MODE != 2) {
        const u16* ap = Abase + kn;
        async_b128(ap,     &As[nxt][lr * 40 + lh]);
        async_b128(ap + 8, &As[nxt][lr * 40 + lh + 8]);
      } else {
        loadAconv(kn, ra0, ra1);
      }
#else
      if constexpr (MODE != 2) {
        const u16* ap = Abase + kn;
        ra0 = *(const u32x4*)ap; ra1 = *(const u32x4*)(ap + 8);
      } else {
        loadAconv(kn, ra0, ra1);
      }
      const u16* bp = Bbase + kn;
      rb0 = *(const u32x4*)bp; rb1 = *(const u32x4*)(bp + 8);
      if (kn + 32 < K) __builtin_prefetch(bp + 32, 0, 1);   // global_prefetch_b8
#endif
    }

    // ---- 8 WMMAs on the current stage
    Frag fa[2], fb[4];
#pragma unroll
    for (int m = 0; m < 2; ++m) {                 // A frag: lane=m-row, K 0-7/16-23 | 8-15/24-31
      int rw = wm * 32 + m * 16 + l15;
      int kh = hi * 8;
      fa[m].q[0] = *(const u32x4*)(&As[cur][rw * 40 + kh]);
      fa[m].q[1] = *(const u32x4*)(&As[cur][rw * 40 + kh + 16]);
    }
#pragma unroll
    for (int n = 0; n < 4; ++n) {                 // B frag: lane=n-col, K 0-7/16-23 | 8-15/24-31
      int rw = wn * 64 + n * 16 + l15;
      int kh = hi * 8;
      fb[n].q[0] = *(const u32x4*)(&Bs[cur][rw * 40 + kh]);
      fb[n].q[1] = *(const u32x4*)(&Bs[cur][rw * 40 + kh + 16]);
    }
#pragma unroll
    for (int m = 0; m < 2; ++m)
#pragma unroll
      for (int n = 0; n < 4; ++n) acc[m][n] = wmma_bf16(fa[m], fb[n], acc[m][n]);

    // ---- commit register-staged pieces of the next stage
    if (more) {
#if ASYNC_LDS
      if constexpr (MODE == 2) {
        *(u32x4*)(&As[nxt][lr * 40 + lh])     = ra0;
        *(u32x4*)(&As[nxt][lr * 40 + lh + 8]) = ra1;
      }
#else
      *(u32x4*)(&As[nxt][lr * 40 + lh])     = ra0;
      *(u32x4*)(&As[nxt][lr * 40 + lh + 8]) = ra1;
      *(u32x4*)(&Bs[nxt][lr * 40 + lh])     = rb0;
      *(u32x4*)(&Bs[nxt][lr * 40 + lh + 8]) = rb1;
#endif
      cur = nxt;
    }
  }

  // ---- epilogue — C layout: VGPR r -> row r (lanes 0-15) / row r+8 (lanes 16-31)
#pragma unroll
  for (int m = 0; m < 2; ++m)
#pragma unroll
    for (int n = 0; n < 4; ++n)
#pragma unroll
      for (int r = 0; r < 8; ++r) {
        int row = bm * 128 + wm * 32 + m * 16 + r + 8 * hi;
        int col = bn * 128 + wn * 64 + n * 16 + l15;
        float v = acc[m][n][r];
        if constexpr (MODE == 0) {
          Cb[(size_t)row * N + col] = f2bf(v);
        } else if constexpr (MODE == 1) {
          Cf[(size_t)row * N + col] = v + R[(size_t)row * N + col];
        } else {
          v = (v + bias[col]) * 0.3333333333f;                  // + conv_b, * 9^-0.5
          v = 0.5f * v * (1.f + erff(v * 0.7071067811865475f)); // exact GeLU
          Cb[(size_t)row * N + col] = f2bf(v);
        }
      }
}

// ---------------------------------------------------------------- fused flash attention
// grid.x = HEADS*BATCH, grid.y = SEQ/128 ; 8 waves, each owns 16 query rows.
__global__ __launch_bounds__(256) void attn_kernel(const u16* __restrict__ qb,
                                                   const u16* __restrict__ kb,
                                                   const u16* __restrict__ vb,
                                                   const float* __restrict__ mask,
                                                   u16* __restrict__ ob) {
  const int hb = blockIdx.x;
  const int b = hb & 3, h = hb >> 2;
  const int wid = threadIdx.x >> 5, lane = threadIdx.x & 31;
  const int l15 = lane & 15, hi = lane >> 4;
  const int s0 = blockIdx.y * 128 + wid * 16;

  __shared__ u16 Vt[64 * 40];        // V tile transposed: [d][t] (stride 40)
  __shared__ u16 Pl[8][16 * 40];     // per-wave P tile: [m][t32]

  v8f zf = {0.f, 0.f, 0.f, 0.f, 0.f, 0.f, 0.f, 0.f};
  v8f o[4]; float mi[8], li[8];
#pragma unroll
  for (int d4 = 0; d4 < 4; ++d4) o[d4] = zf;
#pragma unroll
  for (int r = 0; r < 8; ++r) { mi[r] = -1e30f; li[r] = 0.f; }

  for (int t0 = 0; t0 < SEQ; t0 += 32) {
    __syncthreads();
    {   // cooperative transposed V tile load: 32 t-rows x 64 d
      int idx = threadIdx.x * 8;
      int tl = idx >> 6, d0 = idx & 63;
      u32x4 raw = *(const u32x4*)(vb + (((size_t)(t0 + tl) * BATCH + b) << 10) + h * HD + d0);
      const u16* pr = (const u16*)&raw;
#pragma unroll
      for (int j = 0; j < 8; ++j) Vt[(d0 + j) * 40 + tl] = pr[j];
    }
    __syncthreads();

    // --- scores: S(16 x 32) = q(16 x 64) * k(32 x 64)^T, straight from global bf16
    v8f st[2]; st[0] = zf; st[1] = zf;
#pragma unroll
    for (int dk = 0; dk < HD; dk += 32) {
      Frag fqa;
      {
        int srow = s0 + l15, kh = hi * 8;
        const u16* ap = qb + (((size_t)srow * BATCH + b) << 10) + h * HD + dk + kh;
        fqa.q[0] = *(const u32x4*)ap; fqa.q[1] = *(const u32x4*)(ap + 16);
      }
#pragma unroll
      for (int n0 = 0; n0 < 2; ++n0) {
        Frag fkb;
        int tcol = t0 + n0 * 16 + l15, kh = hi * 8;
        const u16* bp = kb + (((size_t)tcol * BATCH + b) << 10) + h * HD + dk + kh;
        fkb.q[0] = *(const u32x4*)bp; fkb.q[1] = *(const u32x4*)(bp + 16);
        st[n0] = wmma_bf16(fqa, fkb, st[n0]);
      }
    }

    // --- scale * multiplicative mask, online softmax (16-lane row groups)
    float sm[2][8];
#pragma unroll
    for (int n0 = 0; n0 < 2; ++n0)
#pragma unroll
      for (int r = 0; r < 8; ++r) {
        int srow = s0 + r + 8 * hi;
        int tc = t0 + n0 * 16 + l15;
        sm[n0][r] = st[n0][r] * 0.125f * mask[(size_t)srow * SEQ + tc];
      }
    float alpha[8];
#pragma unroll
    for (int r = 0; r < 8; ++r) {
      float tm = fmaxf(sm[0][r], sm[1][r]);
#pragma unroll
      for (int off = 1; off <= 8; off <<= 1) tm = fmaxf(tm, __shfl_xor(tm, off, 32));
      float mn = fmaxf(mi[r], tm);
      alpha[r] = __expf(mi[r] - mn);
      float p0 = __expf(sm[0][r] - mn);
      float p1 = __expf(sm[1][r] - mn);
      sm[0][r] = p0; sm[1][r] = p1;
      float s = p0 + p1;
#pragma unroll
      for (int off = 1; off <= 8; off <<= 1) s += __shfl_xor(s, off, 32);
      li[r] = li[r] * alpha[r] + s;
      mi[r] = mn;
    }
#pragma unroll
    for (int d4 = 0; d4 < 4; ++d4)
#pragma unroll
      for (int r = 0; r < 8; ++r) o[d4][r] *= alpha[r];

    // --- P (C layout) -> LDS -> A fragment layout
    u16* pw = &Pl[wid][0];
#pragma unroll
    for (int n0 = 0; n0 < 2; ++n0)
#pragma unroll
      for (int r = 0; r < 8; ++r)
        pw[(r + 8 * hi) * 40 + n0 * 16 + l15] = f2bf(sm[n0][r]);

    Frag fp;
    {
      int kh = hi * 8;
      fp.q[0] = *(const u32x4*)(pw + l15 * 40 + kh);
      fp.q[1] = *(const u32x4*)(pw + l15 * 40 + kh + 16);
    }
    // --- o(16 x 64) += P(16 x 32) * V(32 x 64)
#pragma unroll
    for (int d4 = 0; d4 < 4; ++d4) {
      Frag fv;
      int drow = d4 * 16 + l15, kh = hi * 8;
      fv.q[0] = *(const u32x4*)(&Vt[drow * 40 + kh]);
      fv.q[1] = *(const u32x4*)(&Vt[drow * 40 + kh + 16]);
      o[d4] = wmma_bf16(fp, fv, o[d4]);
    }
  }

  // --- normalize and emit bf16 attention output (layout (s*B+b, h*64+d))
#pragma unroll
  for (int d4 = 0; d4 < 4; ++d4)
#pragma unroll
    for (int r = 0; r < 8; ++r) {
      int srow = s0 + r + 8 * hi;
      int dcol = h * HD + d4 * 16 + l15;
      ob[(((size_t)srow * BATCH + b) << 10) + dcol] = f2bf(o[d4][r] / li[r]);
    }
}

// ---------------------------------------------------------------- host launcher
extern "C" void kernel_launch(void* const* d_in, const int* in_sizes, int n_in,
                              void* d_out, int out_size, void* d_ws, size_t ws_size,
                              hipStream_t stream) {
  const float* x     = (const float*)d_in[0];
  const float* mask  = (const float*)d_in[1];
  const float* Wq    = (const float*)d_in[2];
  const float* Wk    = (const float*)d_in[3];
  const float* Wv    = (const float*)d_in[4];
  const float* Wo    = (const float*)d_in[5];
  const float* ln1g  = (const float*)d_in[6];
  const float* ln1b  = (const float*)d_in[7];
  const float* ln2g  = (const float*)d_in[8];
  const float* ln2b  = (const float*)d_in[9];
  const float* convw = (const float*)d_in[10];
  const float* convb = (const float*)d_in[11];
  const float* W2    = (const float*)d_in[12];
  float* out = (float*)d_out;

  size_t off = 0;
  auto alloc = [&](size_t bytes) -> void* {
    void* p = (char*)d_ws + off;
    off += (bytes + 255) & ~(size_t)255;
    return p;
  };
  u16* wq_bf = (u16*)alloc((size_t)DIM * DIM * 2);
  u16* wk_bf = (u16*)alloc((size_t)DIM * DIM * 2);
  u16* wv_bf = (u16*)alloc((size_t)DIM * DIM * 2);
  u16* wo_bf = (u16*)alloc((size_t)DIM * DIM * 2);
  u16* w2_bf = (u16*)alloc((size_t)DIM * FILT * 2);
  u16* cw_bf = (u16*)alloc((size_t)FILT * KW * DIM * 2);
  u16* h1    = (u16*)alloc((size_t)NROW * DIM * 2);
  u16* qb    = (u16*)alloc((size_t)NROW * DIM * 2);
  u16* kbuf  = (u16*)alloc((size_t)NROW * DIM * 2);
  u16* vbuf  = (u16*)alloc((size_t)NROW * DIM * 2);
  u16* ab    = (u16*)alloc((size_t)NROW * DIM * 2);
  float* x1  = (float*)alloc((size_t)NROW * DIM * 4);
  u16* h2    = (u16*)alloc((size_t)NROW * DIM * 2);
  u16* yb    = (u16*)alloc((size_t)NROW * FILT * 2);

  const int wsz = DIM * DIM;                    // 1M elems
  cvt_kernel<<<(wsz + 255) / 256, 256, 0, stream>>>(Wq, wq_bf, wsz);
  cvt_kernel<<<(wsz + 255) / 256, 256, 0, stream>>>(Wk, wk_bf, wsz);
  cvt_kernel<<<(wsz + 255) / 256, 256, 0, stream>>>(Wv, wv_bf, wsz);
  cvt_kernel<<<(wsz + 255) / 256, 256, 0, stream>>>(Wo, wo_bf, wsz);
  cvt_kernel<<<(DIM * FILT + 255) / 256, 256, 0, stream>>>(W2, w2_bf, DIM * FILT);
  convw_kernel<<<(FILT * KW * DIM) / 256, 256, 0, stream>>>(convw, cw_bf);

  // attention block
  ln_kernel<<<NROW, 256, 0, stream>>>(x, ln1g, ln1b, h1);
  gemm_nt<0><<<dim3(DIM / 128, NROW / 128), 256, 0, stream>>>(h1, wq_bf, qb, nullptr, nullptr, nullptr, NROW, DIM, DIM);
  gemm_nt<0><<<dim3(DIM / 128, NROW / 128), 256, 0, stream>>>(h1, wk_bf, kbuf, nullptr, nullptr, nullptr, NROW, DIM, DIM);
  gemm_nt<0><<<dim3(DIM / 128, NROW / 128), 256, 0, stream>>>(h1, wv_bf, vbuf, nullptr, nullptr, nullptr, NROW, DIM, DIM);
  attn_kernel<<<dim3(HEADS * BATCH, SEQ / 128), 256, 0, stream>>>(qb, kbuf, vbuf, mask, ab);
  gemm_nt<1><<<dim3(DIM / 128, NROW / 128), 256, 0, stream>>>(ab, wo_bf, nullptr, x1, x, nullptr, NROW, DIM, DIM);

  // conv FFN block
  ln_kernel<<<NROW, 256, 0, stream>>>(x1, ln2g, ln2b, h2);
  gemm_nt<2><<<dim3(FILT / 128, NROW / 128), 256, 0, stream>>>(h2, cw_bf, yb, nullptr, nullptr, convb, NROW, FILT, KW * DIM);
  gemm_nt<1><<<dim3(DIM / 128, NROW / 128), 256, 0, stream>>>(yb, w2_bf, nullptr, out, x1, nullptr, NROW, DIM, FILT);
}
